// RWKV6_ChannelMix_77154792505751
// MI455X (gfx1250) — compile-verified, matrix operand order checked by reading, not run on hardware
//
#include <hip/hip_runtime.h>

typedef __attribute__((ext_vector_type(16))) __bf16 v16bf;
typedef __attribute__((ext_vector_type(8)))  __bf16 v8bf;
typedef __attribute__((ext_vector_type(8)))  float  v8f;

constexpr int  Bb = 4, Tt = 2048, Cc = 4096, Ff = 14336;
constexpr long BT = (long)Bb * Tt;

#define BM 128
#define BN 128
#define BK 32
#define LDK 40   // padded LDS row stride (elements): 80B = 20 banks -> conflict-free b128 reads

// ---------------------------------------------------------------------------
// 16-byte global(bf16) -> LDS async copy (ASYNCcnt-tracked, gfx1250).
// GV mode: global_load_async_to_lds_b128 <lds_addr_vgpr>, <vaddr64>, off
// LDS byte address = low 32 bits of the flat shared pointer (ISA 10.2).
// ---------------------------------------------------------------------------
__device__ inline void copy16_g2l(const __bf16* g, __bf16* l)
{
    unsigned lds_addr = (unsigned)(unsigned long long)l;
    asm volatile("global_load_async_to_lds_b128 %0, %1, off"
                 :
                 : "v"(lds_addr), "v"(g)
                 : "memory");
}

__device__ inline void wait_async_copies()
{
#if __has_builtin(__builtin_amdgcn_s_wait_asynccnt)
    __builtin_amdgcn_s_wait_asynccnt(0);
#else
    asm volatile("s_wait_asynccnt 0x0" ::: "memory");
#endif
}

// ---------------------------------------------------------------------------
// Prep: xk = x + tmk*(shift(x)-x), xr likewise; cast to bf16. new_state = x[:,-1].
// ---------------------------------------------------------------------------
__global__ __launch_bounds__(256) void prep_kernel(
    const float* __restrict__ x, const float* __restrict__ last_state,
    const float* __restrict__ tmk, const float* __restrict__ tmr,
    __bf16* __restrict__ xk, __bf16* __restrict__ xr,
    float* __restrict__ new_state)
{
    long i4 = ((long)blockIdx.x * blockDim.x + threadIdx.x) * 4;
    if (i4 >= BT * (long)Cc) return;
    int  c  = (int)(i4 % Cc);
    long bt = i4 / Cc;
    int  t  = (int)(bt % Tt);
    int  b  = (int)(bt / Tt);

    float4 xv = *(const float4*)(x + i4);
    float4 xx;
    if (t == 0) xx = *(const float4*)(last_state + (long)b * Cc + c);
    else        xx = *(const float4*)(x + i4 - Cc);
    float4 mk = *(const float4*)(tmk + c);
    float4 mr = *(const float4*)(tmr + c);

    __bf16 ok[4], orr[4];
    ok[0]  = (__bf16)(xv.x + mk.x * (xx.x - xv.x));
    ok[1]  = (__bf16)(xv.y + mk.y * (xx.y - xv.y));
    ok[2]  = (__bf16)(xv.z + mk.z * (xx.z - xv.z));
    ok[3]  = (__bf16)(xv.w + mk.w * (xx.w - xv.w));
    orr[0] = (__bf16)(xv.x + mr.x * (xx.x - xv.x));
    orr[1] = (__bf16)(xv.y + mr.y * (xx.y - xv.y));
    orr[2] = (__bf16)(xv.z + mr.z * (xx.z - xv.z));
    orr[3] = (__bf16)(xv.w + mr.w * (xx.w - xv.w));
    *(uint2*)(xk + i4) = *(const uint2*)ok;
    *(uint2*)(xr + i4) = *(const uint2*)orr;

    if (t == Tt - 1) *(float4*)(new_state + (long)b * Cc + c) = xv;
}

// ---------------------------------------------------------------------------
// Load one 16x32 (MxK) bf16 fragment from an LDS tile with padded ld = LDK.
// ISA 7.12.2 layout: lane<16 -> row=lane, K 0..7 & 16..23;
//                    lane>=16 -> row=lane-16, K 8..15 & 24..31.
// ---------------------------------------------------------------------------
__device__ inline v16bf lds_frag(const __bf16* p, int lane)
{
    int r    = lane & 15;
    int koff = (lane >> 4) << 3;          // 0 or 8
    const __bf16* q = p + r * LDK + koff;
    v8bf lo = *(const v8bf*)(q);
    v8bf hi = *(const v8bf*)(q + 16);
    v16bf o;
#pragma unroll
    for (int i = 0; i < 8; ++i) { o[i] = lo[i]; o[i + 8] = hi[i]; }
    return o;
}

// ---------------------------------------------------------------------------
// D = A(MxK bf16) x W(NxK f32, K-contiguous)^T; bf16 WMMA, f32 accumulate.
// Double-buffered LDS software pipeline; A tile via async global->LDS copies.
// EPI 0: relu(d)^2 -> bf16 outBF     (k path)
// EPI 1: sigmoid(d) -> f32 outF      (r path)
// EPI 2: outF *= d                   (kv path, fused r*kv)
// ---------------------------------------------------------------------------
template <int EPI>
__global__ __launch_bounds__(256) void gemm_wmma(
    const __bf16* __restrict__ A, const float* __restrict__ W,
    float* __restrict__ outF, __bf16* __restrict__ outBF,
    int K, int ldOut)
{
    __shared__ __align__(16) __bf16 sA[2][BM * LDK];
    __shared__ __align__(16) __bf16 sB[2][BN * LDK];

    const int tid  = threadIdx.x;
    const int lane = tid & 31;
    const int wave = tid >> 5;
    const int wm   = wave & 1;            // 2 waves along M
    const int wn   = wave >> 1;           // 4 waves along N
    const long mBase = (long)blockIdx.y * BM;
    const long nBase = (long)blockIdx.x * BN;

    v8f acc[4][2];
#pragma unroll
    for (int mi = 0; mi < 4; ++mi)
#pragma unroll
        for (int ni = 0; ni < 2; ++ni)
#pragma unroll
            for (int v = 0; v < 8; ++v) acc[mi][ni][v] = 0.0f;

    const int row  = tid >> 1;            // 0..127
    const int half = (tid & 1) * 16;      // 0 or 16 (K columns)
    const __bf16* gA = A + (mBase + row) * (long)K + half;
    const float*  gW = W + (nBase + row) * (long)K + half;

    auto stageA = [&](int k0, int buf) {
        const __bf16* src = gA + k0;
        __bf16*       dst = &sA[buf][row * LDK + half];
        copy16_g2l(src,     dst);
        copy16_g2l(src + 8, dst + 8);
    };
    auto loadW = [&](int k0, float* f) {
        const float* src = gW + k0;
#pragma unroll
        for (int i = 0; i < 4; ++i)
            *(float4*)(f + 4 * i) = *(const float4*)(src + 4 * i);
    };
    auto storeW = [&](int buf, const float* f) {
        __bf16 h[16];
#pragma unroll
        for (int i = 0; i < 16; ++i) h[i] = (__bf16)f[i];
        __bf16* dst = &sB[buf][row * LDK + half];
        *(uint4*)(dst)     = *(const uint4*)(h);
        *(uint4*)(dst + 8) = *(const uint4*)(h + 8);
    };

    // ---- prologue: stage tile 0 into buffer 0 ----
    {
        stageA(0, 0);
        float f[16];
        loadW(0, f);
        storeW(0, f);
    }
    wait_async_copies();
    __syncthreads();

    const int nIter = K / BK;
    for (int it = 0; it < nIter; ++it) {
        const int cur = it & 1;
        const int nxt = cur ^ 1;
        const int k0  = it * BK;
        const bool more = (it + 1 < nIter);

        // ---- issue next tile's global traffic before computing ----
        float fw[16];
        if (more) {
            stageA(k0 + BK, nxt);           // async bf16 A -> LDS[nxt]
            loadW(k0 + BK, fw);             // f32 W -> regs
            if (it + 2 < nIter)
                __builtin_prefetch(gW + k0 + 2 * BK, 0, 1);  // global_prefetch_b8
        }

        // ---- compute on current buffer ----
        v16bf af[4], bfb[2];
#pragma unroll
        for (int mi = 0; mi < 4; ++mi)
            af[mi] = lds_frag(&sA[cur][(wm * 64 + mi * 16) * LDK], lane);
#pragma unroll
        for (int ni = 0; ni < 2; ++ni)
            bfb[ni] = lds_frag(&sB[cur][(wn * 32 + ni * 16) * LDK], lane);

#pragma unroll
        for (int mi = 0; mi < 4; ++mi)
#pragma unroll
            for (int ni = 0; ni < 2; ++ni)
                acc[mi][ni] = __builtin_amdgcn_wmma_f32_16x16x32_bf16(
                    false, af[mi], false, bfb[ni],
                    (short)0, acc[mi][ni], false, false);

        // ---- finish staging next buffer, then one barrier per iteration ----
        if (more) {
            storeW(nxt, fw);                // cvt_pk_bf16 + ds_store
            wait_async_copies();
        }
        __syncthreads();
    }

    // ---- epilogue (C/D layout: lane<16 -> M=v, lane>=16 -> M=v+8; N=lane&15) ----
    const int mlane = (lane >> 4) * 8;
    const int nlane = lane & 15;
#pragma unroll
    for (int mi = 0; mi < 4; ++mi) {
#pragma unroll
        for (int ni = 0; ni < 2; ++ni) {
#pragma unroll
            for (int v = 0; v < 8; ++v) {
                long m = mBase + wm * 64 + mi * 16 + v + mlane;
                long n = nBase + wn * 32 + ni * 16 + nlane;
                float d = acc[mi][ni][v];
                if (EPI == 0) {
                    d = fmaxf(d, 0.0f);
                    outBF[m * ldOut + n] = (__bf16)(d * d);
                } else if (EPI == 1) {
                    outF[m * ldOut + n] = 1.0f / (1.0f + __expf(-d));
                } else {
                    long idx = m * ldOut + n;
                    outF[idx] = outF[idx] * d;
                }
            }
        }
    }
}

// ---------------------------------------------------------------------------
extern "C" void kernel_launch(void* const* d_in, const int* in_sizes, int n_in,
                              void* d_out, int out_size, void* d_ws, size_t ws_size,
                              hipStream_t stream)
{
    const float* x          = (const float*)d_in[0];
    const float* last_state = (const float*)d_in[1];
    const float* tmk        = (const float*)d_in[2];
    const float* tmr        = (const float*)d_in[3];
    const float* key_w      = (const float*)d_in[4];
    const float* recep_w    = (const float*)d_in[5];
    const float* value_w    = (const float*)d_in[6];

    float* out       = (float*)d_out;
    float* new_state = out + BT * (long)Cc;

    __bf16* xk = (__bf16*)d_ws;
    __bf16* xr = xk + BT * (long)Cc;
    __bf16* kk = xr + BT * (long)Cc;   // relu(k)^2 in bf16, BT x F

    // 1) token-shift lerps + bf16 cast + new_state
    long n4 = BT * (long)Cc / 4;
    prep_kernel<<<(unsigned)((n4 + 255) / 256), 256, 0, stream>>>(
        x, last_state, tmk, tmr, xk, xr, new_state);

    // 2) kk = relu(xk @ key_w^T)^2            (M=8192, N=14336, K=4096)
    dim3 g1(Ff / BN, (unsigned)(BT / BM));
    gemm_wmma<0><<<g1, 256, 0, stream>>>(xk, key_w, nullptr, kk, Cc, Ff);

    // 3) out = sigmoid(xr @ receptance_w^T)   (M=8192, N=4096, K=4096)
    dim3 g2(Cc / BN, (unsigned)(BT / BM));
    gemm_wmma<1><<<g2, 256, 0, stream>>>(xr, recep_w, out, nullptr, Cc, Cc);

    // 4) out *= kk @ value_w^T                (M=8192, N=4096, K=14336)
    gemm_wmma<2><<<g2, 256, 0, stream>>>(kk, value_w, out, nullptr, Ff, Cc);
}
